// MyModel_87522843560143
// MI455X (gfx1250) — matrix operand order repaired and checked by reading, not compiled, not run on hardware
//
#include <hip/hip_runtime.h>

typedef __attribute__((ext_vector_type(16))) _Float16 v16h;
typedef __attribute__((ext_vector_type(8)))  _Float16 v8h;
typedef __attribute__((ext_vector_type(8)))  float    v8f;
typedef __attribute__((ext_vector_type(4)))  float    v4f;

// global (address_space(1)) pointer types so weight loads are guaranteed
// global_load_b128 (not flat_load: FLAT ticks DScnt too and ties up the LDS path)
typedef const char __attribute__((address_space(1)))* gchar_cp;
typedef const v16h __attribute__((address_space(1)))* gv16h_cp;

namespace {
constexpr int kB = 64, kT = 1024, kF = 128, kU = 192, kG = 768, kO = 64;

// ---- workspace layout (in _Float16 units) ----
// Per layer, stacked weights [W;U] (K x 768) stored as WMMA B-fragments:
// tiles ordered [nt][kt], each tile = 32 lanes x 16 halves (1024 bytes).
// B-fragment k order chosen here (contiguous, same as A): element j of lane l:
//   N = nt*16 + (l&15),  K = kt*32 + j + (l>=16 ? 16 : 0)
constexpr int    L0_KT  = (kF + kU) / 32;   // 10  (A = [x | h0], K=320)
constexpr int    L12_KT = (kU + kU) / 32;   // 12  (A = [h | h], K=384)
constexpr int    WD_KT  = kU / 32;          // 6
constexpr size_t L0_OFF = 0;
constexpr size_t L1_OFF = L0_OFF + (size_t)L0_KT  * 48 * 512;
constexpr size_t L2_OFF = L1_OFF + (size_t)L12_KT * 48 * 512;
constexpr size_t WD_OFF = L2_OFF + (size_t)L12_KT * 48 * 512;
constexpr size_t WS_HALVES = WD_OFF + (size_t)WD_KT * 4 * 512;  // 847,872 (~1.62 MB)

// ---- LDS layout ----
// halves:  x panel [64][128], then 6 h panels [64][192] (3 states x 2 time-parity bufs)
// floats:  c state [3][64][192] appended after the halves
constexpr int X_OFF   = 0;                       // stride 128 halves
constexpr int H_BASE  = kB * kF;                 // 8192 halves
constexpr int H_PANEL = kB * kU;                 // 12288 halves per panel
constexpr int C_HALF  = H_BASE + 6 * H_PANEL;    // 81920 halves = 163,840 B
constexpr size_t LDS_BYTES = (size_t)C_HALF * 2 + (size_t)3 * kB * kU * 4;  // 311,296 B
}  // namespace

// ---------------- prep: f32 weights -> f16 swizzled B-fragments ----------------
__global__ void lstm_prep_kernel(const float* __restrict__ W0, const float* __restrict__ U0,
                                 const float* __restrict__ W1, const float* __restrict__ U1,
                                 const float* __restrict__ W2, const float* __restrict__ U2,
                                 const float* __restrict__ Wd, _Float16* __restrict__ ws) {
  for (size_t id = (size_t)blockIdx.x * blockDim.x + threadIdx.x; id < WS_HALVES;
       id += (size_t)gridDim.x * blockDim.x) {
    if (id >= WD_OFF) {  // dense Wd [192 x 64]
      size_t rel = id - WD_OFF;
      int tile = (int)(rel >> 9), within = (int)(rel & 511);
      int lane = within >> 4, j = within & 15;
      int nt = tile / WD_KT, kt = tile % WD_KT;
      int n = nt * 16 + (lane & 15);
      int k = kt * 32 + j + ((lane >= 16) ? 16 : 0);
      ws[id] = (_Float16)Wd[k * kO + n];
      continue;
    }
    size_t rel; int nkt, frows; const float *Wm, *Um;
    if (id < L1_OFF)      { rel = id - L0_OFF; nkt = L0_KT;  frows = kF; Wm = W0; Um = U0; }
    else if (id < L2_OFF) { rel = id - L1_OFF; nkt = L12_KT; frows = kU; Wm = W1; Um = U1; }
    else                  { rel = id - L2_OFF; nkt = L12_KT; frows = kU; Wm = W2; Um = U2; }
    int tile = (int)(rel >> 9), within = (int)(rel & 511);
    int lane = within >> 4, j = within & 15;
    int nt = tile / nkt, kt = tile % nkt;
    int n = nt * 16 + (lane & 15);
    int k = kt * 32 + j + ((lane >= 16) ? 16 : 0);
    float v = (k < frows) ? Wm[(size_t)k * kG + n] : Um[(size_t)(k - frows) * kG + n];
    ws[id] = (_Float16)v;
  }
}

// ---------------- fragment helpers ----------------
// A-fragment, contiguous k order used here: lane l<16: M=l, K={0..15}; lane l>=16: M=l-16,
// K={16..31}.  From row-major LDS: two 16-byte ds_load_b128 per lane.
__device__ __forceinline__ v16h load_a_frag(const _Float16* panel, int stride,
                                            int mtile, int kk, int lhalf, int hi) {
  const _Float16* p = panel + (mtile * 16 + lhalf) * stride + kk * 32 + hi * 16;
  v8h lo = *(const v8h*)p;
  v8h hh = *(const v8h*)(p + 8);
  return __builtin_shufflevector(lo, hh, 0, 1, 2, 3, 4, 5, 6, 7, 8, 9, 10, 11, 12, 13, 14, 15);
}

// B-fragment: uniform SGPR64 base + unsigned 32-bit per-lane byte offset ->
// backend selects global_load_b128 vdst, voffset, s[base] (no 64-bit VALU math).
__device__ __forceinline__ v16h load_b_frag(const _Float16* wsl, unsigned byteOff) {
  gv16h_cp p = (gv16h_cp)((gchar_cp)wsl + byteOff);
  return *p;
}

__device__ __forceinline__ float sigm(float x) {
  return __builtin_amdgcn_rcpf(1.0f + __expf(-x));
}
__device__ __forceinline__ float tanh_fast(float x) {
  float e = __expf(2.0f * x);
  return 1.0f - 2.0f * __builtin_amdgcn_rcpf(e + 1.0f);
}

// ---------------- persistent sequential LSTM kernel (1 workgroup, 16 waves) ----------------
__global__ void __launch_bounds__(512, 1)
lstm_seq_kernel(const float* __restrict__ x, const _Float16* __restrict__ ws,
                const float* __restrict__ b0, const float* __restrict__ b1,
                const float* __restrict__ b2, const float* __restrict__ bd,
                const float* __restrict__ h_init, const float* __restrict__ c_init,
                float* __restrict__ out) {
  extern __shared__ _Float16 lds[];
  float* cf = (float*)(lds + C_HALF);  // cell state, f32, [3][64][192]
  const int tid = threadIdx.x;
  const int wave = tid >> 5;  // 16 waves (wave32)
  const int lane = tid & 31;
  const int lhalf = lane & 15;
  const int hi = lane >> 4;

  // ---- init: h goes into parity-1 buffers (read at t=0), c copied as-is ----
  for (int i = tid; i < 3 * kB * kU; i += 512) {
    int lay = i / (kB * kU);
    int rc = i % (kB * kU);
    lds[H_BASE + (lay * 2 + 1) * H_PANEL + rc] = (_Float16)h_init[i];
    cf[i] = c_init[i];
  }
  __syncthreads();

  const float* bptrA[3] = {b0, b1, b2};
  const int    nktA[3]  = {L0_KT, L12_KT, L12_KT};
  const size_t wsOffA[3] = {L0_OFF, L1_OFF, L2_OFF};

  for (int t = 0; t < kT; ++t) {
    const int buf = t & 1, pbuf = buf ^ 1;

    // Launder a zero offset to block LICM of weight loads across the t loop, then
    // re-uniformize with readfirstlane so the base stays an SGPR (uniform) and the
    // backend can use the scalar-base + voffset global addressing mode.
    int lz = 0;
    asm volatile("" : "+s"(lz));
    lz = __builtin_amdgcn_readfirstlane(lz);
    const _Float16* wsp = ws + lz;

    // ---- stage x_t (f32 -> f16), 16 floats per thread ----
    {
      int b = tid >> 3;
      int f = (tid & 7) * 16;
      const float* xp = x + ((size_t)b * kT + t) * kF + f;
      v4f a0 = *(const v4f*)(xp);
      v4f a1 = *(const v4f*)(xp + 4);
      v4f a2 = *(const v4f*)(xp + 8);
      v4f a3 = *(const v4f*)(xp + 12);
      v8h o0, o1;
#pragma unroll
      for (int j = 0; j < 4; ++j) {
        o0[j] = (_Float16)a0[j]; o0[j + 4] = (_Float16)a1[j];
        o1[j] = (_Float16)a2[j]; o1[j + 4] = (_Float16)a3[j];
      }
      *(v8h*)(lds + X_OFF + b * kF + f) = o0;
      *(v8h*)(lds + X_OFF + b * kF + f + 8) = o1;
    }
    __syncthreads();

#pragma unroll
    for (int lay = 0; lay < 3; ++lay) {
      const _Float16* wsl = wsp + wsOffA[lay];
      const int nkt = nktA[lay];
      // A-panel parts: [input | recurrent]
      const _Float16* part0 = (lay == 0) ? (lds + X_OFF)
                                         : (lds + H_BASE + ((lay - 1) * 2 + buf) * H_PANEL);
      const int s0 = (lay == 0) ? kF : kU;
      const int k0 = (lay == 0) ? 4 : 6;
      const _Float16* part1 = lds + H_BASE + (lay * 2 + pbuf) * H_PANEL;
      _Float16* hout = lds + H_BASE + (lay * 2 + buf) * H_PANEL;
      const float* bl = bptrA[lay];

#pragma unroll 1  // keep pairs sequential: bounds live registers, blocks hoisting
      for (int p = 0; p < 3; ++p) {
        const int q = wave * 3 + p;
        const int m = q / 12, u = q % 12;
        const unsigned laneB = (unsigned)lane * 32u;

        v8f acc[4];
#pragma unroll
        for (int g = 0; g < 4; ++g) {
          float bv = bl[g * kU + u * 16 + lhalf];  // bias folded into accumulator
#pragma unroll
          for (int r = 0; r < 8; ++r) acc[g][r] = bv;
        }

        int ktg = 0;
#pragma unroll 2
        for (int kk = 0; kk < k0; ++kk, ++ktg) {
          v16h a = load_a_frag(part0, s0, m, kk, lhalf, hi);
#pragma unroll
          for (int g = 0; g < 4; ++g) {
            unsigned off = ((unsigned)((g * 12 + u) * nkt + ktg) << 10) + laneB;
            v16h bb = load_b_frag(wsl, off);
            acc[g] = __builtin_amdgcn_wmma_f32_16x16x32_f16(
                false, a, false, bb, (short)0, acc[g], false, false);
          }
        }
#pragma unroll 2
        for (int kk = 0; kk < 6; ++kk, ++ktg) {
          v16h a = load_a_frag(part1, kU, m, kk, lhalf, hi);
#pragma unroll
          for (int g = 0; g < 4; ++g) {
            unsigned off = ((unsigned)((g * 12 + u) * nkt + ktg) << 10) + laneB;
            v16h bb = load_b_frag(wsl, off);
            acc[g] = __builtin_amdgcn_wmma_f32_16x16x32_f16(
                false, a, false, bb, (short)0, acc[g], false, false);
          }
        }

        // gates; c lives in LDS (each lane exclusively owns its elements)
#pragma unroll
        for (int r = 0; r < 8; ++r) {
          int row = m * 16 + r + hi * 8;  // C-fragment M mapping
          int cidx = (lay * kB + row) * kU + u * 16 + lhalf;
          float iv = sigm(acc[0][r]);
          float fv = sigm(acc[1][r]);
          float gv = tanh_fast(acc[2][r]);
          float ov = sigm(acc[3][r]);
          float cv = fv * cf[cidx] + iv * gv;
          cf[cidx] = cv;
          hout[row * kU + u * 16 + lhalf] = (_Float16)(ov * tanh_fast(cv));
        }
      }
      __syncthreads();  // h(lay,buf) complete before next layer consumes it
    }
  }

  // ---- dense head: sigmoid(h2_last @ Wd + bd); final h2 is in buf = (kT-1)&1 = 1 ----
  {
    const _Float16* h2 = lds + H_BASE + (2 * 2 + 1) * H_PANEL;
    int m = wave >> 2, nt = wave & 3;
    v8f acc;
    float bv = bd[nt * 16 + lhalf];
#pragma unroll
    for (int r = 0; r < 8; ++r) acc[r] = bv;
#pragma unroll 2
    for (int kt = 0; kt < WD_KT; ++kt) {
      v16h a = load_a_frag(h2, kU, m, kt, lhalf, hi);
      unsigned off = ((unsigned)(nt * WD_KT + kt) << 10) + (unsigned)lane * 32u;
      v16h bb = load_b_frag(ws + WD_OFF, off);
      acc = __builtin_amdgcn_wmma_f32_16x16x32_f16(
          false, a, false, bb, (short)0, acc, false, false);
    }
#pragma unroll
    for (int r = 0; r < 8; ++r) {
      int row = m * 16 + r + hi * 8;
      out[(size_t)row * kO + nt * 16 + lhalf] = sigm(acc[r]);
    }
  }
}

extern "C" void kernel_launch(void* const* d_in, const int* in_sizes, int n_in,
                              void* d_out, int out_size, void* d_ws, size_t ws_size,
                              hipStream_t stream) {
  (void)in_sizes; (void)n_in; (void)out_size; (void)ws_size;
  const float* x      = (const float*)d_in[0];
  const float* W0     = (const float*)d_in[1];
  const float* U0     = (const float*)d_in[2];
  const float* b0     = (const float*)d_in[3];
  const float* W1     = (const float*)d_in[4];
  const float* U1     = (const float*)d_in[5];
  const float* b1     = (const float*)d_in[6];
  const float* W2     = (const float*)d_in[7];
  const float* U2     = (const float*)d_in[8];
  const float* b2     = (const float*)d_in[9];
  const float* Wd     = (const float*)d_in[10];
  const float* bd     = (const float*)d_in[11];
  const float* h_init = (const float*)d_in[12];
  const float* c_init = (const float*)d_in[13];
  _Float16* ws = (_Float16*)d_ws;

  // Parallel weight convert/swizzle (whole GPU), then persistent sequential kernel (1 WGP).
  lstm_prep_kernel<<<dim3(1024), dim3(256), 0, stream>>>(W0, U0, W1, U1, W2, U2, Wd, ws);
  lstm_seq_kernel<<<dim3(1), dim3(512), LDS_BYTES, stream>>>(
      x, ws, b0, b1, b2, bd, h_init, c_init, (float*)d_out);
}
